// NLPModel_84241488544450
// MI455X (gfx1250) — matrix operand order repaired and checked by reading, not compiled, not run on hardware
//
#include <hip/hip_runtime.h>
#include <cstdint>

// ---------------- problem constants ----------------
#define Bdim 2
#define Sdim 2048
#define Hdim 512
#define Ldim 3
#define Vdim 32000
#define Mrows (Bdim * Sdim)   // 4096
#define G3H   (3 * Hdim)      // 1536
#define PCH   32              // scan chunks per channel
#define CLEN  (Sdim / PCH)    // 64

typedef __attribute__((ext_vector_type(16))) __bf16 v16bf;
typedef __attribute__((ext_vector_type(8)))  float  v8f;

union Frag { v16bf v; uint4 q[2]; };

// ---------------- bf16 split helpers (hi + lo decomposition) ----------------
__device__ __forceinline__ unsigned short f2bf_bits(float x) {
  unsigned u = __builtin_bit_cast(unsigned, x);
  u += 0x7FFFu + ((u >> 16) & 1u);           // round-to-nearest-even
  return (unsigned short)(u >> 16);
}
__device__ __forceinline__ float bfbits2f(unsigned short s) {
  unsigned u = ((unsigned)s) << 16;
  return __builtin_bit_cast(float, u);
}
__device__ __forceinline__ void split2(float x, unsigned short* hi, unsigned short* lo) {
  unsigned short hb = f2bf_bits(x);
  *hi = hb;
  *lo = f2bf_bits(x - bfbits2f(hb));
}

// ---------------- CDNA5 async global->LDS copy (16B per lane) ----------------
__device__ __forceinline__ void async_copy16(const unsigned short* g, unsigned short* l) {
  unsigned lds_off = (unsigned)(uintptr_t)l;   // low 32 bits of generic LDS addr = LDS offset
  asm volatile("global_load_async_to_lds_b128 %0, %1, off"
               :: "v"(lds_off), "v"(g) : "memory");
}
__device__ __forceinline__ void wait_async0() {
  asm volatile("s_wait_asynccnt 0" ::: "memory");
}

// ---------------- elementwise: split fp32 weights into bf16 hi/lo ----------------
__global__ __launch_bounds__(256) void split_kernel(const float* __restrict__ src,
                                                    unsigned short* __restrict__ hi,
                                                    unsigned short* __restrict__ lo,
                                                    long long n) {
  long long i = (long long)blockIdx.x * blockDim.x + threadIdx.x;
  long long stride = (long long)gridDim.x * blockDim.x;
  for (; i < n; i += stride) split2(src[i], hi + i, lo + i);
}

// ---------------- embedding gather ----------------
__global__ __launch_bounds__(256) void embed_kernel(const int* __restrict__ ids,
                                                    const float* __restrict__ emb,
                                                    float* __restrict__ x) {
  int row = blockIdx.x, t = threadIdx.x;
  int id = ids[row];
  const float* e = emb + (size_t)id * Hdim;
  float* xr = x + (size_t)row * Hdim;
  xr[t] = e[t];
  xr[t + 256] = e[t + 256];
}

// ---------------- LayerNorm (+optional bias) fused with bf16 hi/lo split ----------------
__global__ __launch_bounds__(256) void ln_split_kernel(const float* __restrict__ x,
                                                       const float* __restrict__ w,
                                                       const float* __restrict__ b,
                                                       unsigned short* __restrict__ hi,
                                                       unsigned short* __restrict__ lo) {
  __shared__ float ssum[256], ssq[256];
  int row = blockIdx.x, t = threadIdx.x;
  const float* xr = x + (size_t)row * Hdim;
  float v0 = xr[t], v1 = xr[t + 256];
  ssum[t] = v0 + v1;
  ssq[t]  = v0 * v0 + v1 * v1;
  __syncthreads();
  for (int off = 128; off > 0; off >>= 1) {
    if (t < off) { ssum[t] += ssum[t + off]; ssq[t] += ssq[t + off]; }
    __syncthreads();
  }
  float mean = ssum[0] * (1.0f / Hdim);
  float var  = ssq[0] * (1.0f / Hdim) - mean * mean;
  float inv  = rsqrtf(var + 1e-5f);
  size_t o = (size_t)row * Hdim;
  float y0 = (v0 - mean) * inv * w[t]       + (b ? b[t]       : 0.f);
  float y1 = (v1 - mean) * inv * w[t + 256] + (b ? b[t + 256] : 0.f);
  split2(y0, hi + o + t,       lo + o + t);
  split2(y1, hi + o + t + 256, lo + o + t + 256);
}

// ---------------- WMMA GEMM: C[M,N] = A[M,K] * W[N,K]^T + bias ----------------
// LDS-staged, double-buffered with async global->LDS copies (ASYNCcnt).
// Block = (32,4): 4 waves; block tile = 128(M) x 64(N); wave tile = 32(M) x 64(N).
// Per 32-wide k-step: A tile 128x32 hi+lo (16KB), B tile 64x32 hi+lo (8KB).
__device__ __forceinline__ void fill_tiles(
    const unsigned short* __restrict__ Ahi, const unsigned short* __restrict__ Alo,
    const unsigned short* __restrict__ Whi, const unsigned short* __restrict__ Wlo,
    unsigned short* sAh, unsigned short* sAl,
    unsigned short* sBh, unsigned short* sBl,
    int mBlock, int nBase, int k0, int K, int tid) {
#pragma unroll
  for (int i = 0; i < 4; ++i) {           // A: 512 x 16B segments, 4 per thread per buf
    int seg = tid + i * 128;
    int row = seg >> 2;                   // 0..127
    int cs  = (seg & 3) << 3;             // 0,8,16,24 (elements)
    size_t gb = (size_t)(mBlock + row) * K + k0 + cs;
    int lo = row * 32 + cs;
    async_copy16(Ahi + gb, sAh + lo);
    async_copy16(Alo + gb, sAl + lo);
  }
#pragma unroll
  for (int i = 0; i < 2; ++i) {           // B: 256 x 16B segments, 2 per thread per buf
    int seg = tid + i * 128;
    int row = seg >> 2;                   // 0..63
    int cs  = (seg & 3) << 3;
    size_t gb = (size_t)(nBase + row) * K + k0 + cs;
    int lo = row * 32 + cs;
    async_copy16(Whi + gb, sBh + lo);
    async_copy16(Wlo + gb, sBl + lo);
  }
}

__global__ __launch_bounds__(128) void gemm_bf16x2_wmma(
    const unsigned short* __restrict__ Ahi, const unsigned short* __restrict__ Alo,
    const unsigned short* __restrict__ Whi, const unsigned short* __restrict__ Wlo,
    const float* __restrict__ bias, float* __restrict__ C,
    int M, int N, int K) {
  __shared__ __align__(16) unsigned short sA[2][2][128 * 32];  // [buf][hi/lo] 8KB each
  __shared__ __align__(16) unsigned short sB[2][2][64 * 32];   // [buf][hi/lo] 4KB each

  const int lane  = threadIdx.x;        // 0..31
  const int wave  = threadIdx.y;        // 0..3
  const int tid   = wave * 32 + lane;
  const int lmod  = lane & 15;
  const int lhi   = lane >> 4;          // half-wave select
  const int mBlock = blockIdx.y * 128;
  const int nBase  = blockIdx.x * 64;

  v8f zero = {0.f, 0.f, 0.f, 0.f, 0.f, 0.f, 0.f, 0.f};
  v8f acc[2][4];
#pragma unroll
  for (int i = 0; i < 2; ++i)
#pragma unroll
    for (int j = 0; j < 4; ++j) acc[i][j] = zero;

  const int aOff = lhi * 8;    // A frag: K offset of this half-wave's 8-elem chunks
  const int bOff = lhi * 8;    // B frag: K offset of this half-wave's 8-elem chunks

  const int nk = K / 32;
  fill_tiles(Ahi, Alo, Whi, Wlo, sA[0][0], sA[0][1], sB[0][0], sB[0][1],
             mBlock, nBase, 0, K, tid);
  int cur = 0;

  for (int step = 0; step < nk; ++step) {
    wait_async0();          // this wave's async fills of buffer `cur` are in LDS
    __syncthreads();        // all waves' fills done; all done reading buffer cur^1
    if (step + 1 < nk)
      fill_tiles(Ahi, Alo, Whi, Wlo,
                 sA[cur ^ 1][0], sA[cur ^ 1][1], sB[cur ^ 1][0], sB[cur ^ 1][1],
                 mBlock, nBase, (step + 1) * 32, K, tid);

    Frag ah[2], al[2];
#pragma unroll
    for (int mi = 0; mi < 2; ++mi) {
      const unsigned short* pa = &sA[cur][0][(wave * 32 + mi * 16 + lmod) * 32 + aOff];
      const unsigned short* pl = &sA[cur][1][(wave * 32 + mi * 16 + lmod) * 32 + aOff];
      ah[mi].q[0] = *(const uint4*)pa;
      ah[mi].q[1] = *(const uint4*)(pa + 16);
      al[mi].q[0] = *(const uint4*)pl;
      al[mi].q[1] = *(const uint4*)(pl + 16);
    }
    Frag bh[4], bl[4];
#pragma unroll
    for (int ni = 0; ni < 4; ++ni) {
      const unsigned short* pb = &sB[cur][0][(ni * 16 + lmod) * 32 + bOff];
      const unsigned short* pl = &sB[cur][1][(ni * 16 + lmod) * 32 + bOff];
      bh[ni].q[0] = *(const uint4*)pb;
      bh[ni].q[1] = *(const uint4*)(pb + 16);
      bl[ni].q[0] = *(const uint4*)pl;
      bl[ni].q[1] = *(const uint4*)(pl + 16);
    }
#pragma unroll
    for (int mi = 0; mi < 2; ++mi)
#pragma unroll
      for (int ni = 0; ni < 4; ++ni) {
        acc[mi][ni] = __builtin_amdgcn_wmma_f32_16x16x32_bf16(
            false, ah[mi].v, false, bh[ni].v, (short)0, acc[mi][ni], false, false);
        acc[mi][ni] = __builtin_amdgcn_wmma_f32_16x16x32_bf16(
            false, ah[mi].v, false, bl[ni].v, (short)0, acc[mi][ni], false, false);
        acc[mi][ni] = __builtin_amdgcn_wmma_f32_16x16x32_bf16(
            false, al[mi].v, false, bh[ni].v, (short)0, acc[mi][ni], false, false);
      }
    __syncthreads();        // done reading buffer cur before it is refilled next iter
    cur ^= 1;
  }

#pragma unroll
  for (int mi = 0; mi < 2; ++mi)
#pragma unroll
    for (int ni = 0; ni < 4; ++ni) {
      int n = nBase + ni * 16 + lmod;
      float bv = bias ? bias[n] : 0.f;
#pragma unroll
      for (int r = 0; r < 8; ++r) {
        int m = mBlock + wave * 32 + mi * 16 + lhi * 8 + r;
        C[(size_t)m * N + n] = acc[mi][ni][r] + bv;
      }
    }
}

// ---------------- gate math: log forget / log(i * g(tilde)) ----------------
__device__ __forceinline__ float softplusf_(float x) {
  return fmaxf(x, 0.f) + log1pf(expf(-fabsf(x)));
}
__device__ __forceinline__ float logaddexpf_(float a, float b) {
  float m = fmaxf(a, b), n = fminf(a, b);
  if (m == -__builtin_inff()) return m;
  return m + log1pf(expf(n - m));
}

__global__ __launch_bounds__(256) void gatelog_kernel(const float* __restrict__ gates,
                                                      float* __restrict__ lf,
                                                      float* __restrict__ lv) {
  int idx = blockIdx.x * 256 + threadIdx.x;   // < Mrows*Hdim exactly
  int row = idx >> 9;
  int h   = idx & 511;
  size_t g = (size_t)row * G3H + h;
  float fg = gates[g], ig = gates[g + Hdim], tg = gates[g + 2 * Hdim];
  float diff  = softplusf_(-fg) - softplusf_(-ig);
  float log_f = -softplusf_(diff);
  float log_i = -softplusf_(-diff);
  float log_g = (tg >= 0.f) ? logf(tg + 0.5f) : -softplusf_(-tg);
  lf[idx] = log_f;
  lv[idx] = log_i + log_g;
}

// ---------------- 3-pass chunked log-space scan ----------------
__global__ __launch_bounds__(256) void scan_chunk_kernel(const float* __restrict__ lf,
                                                         const float* __restrict__ lv,
                                                         float* __restrict__ cA,
                                                         float* __restrict__ cV) {
  int tid = blockIdx.x * 256 + threadIdx.x;   // < Bdim*PCH*Hdim = 32768
  int h = tid & 511, p = (tid >> 9) & 31, b = tid >> 14;
  size_t base = ((size_t)(b * Sdim + p * CLEN)) * Hdim + h;
  float A = 0.f, Vv = -__builtin_inff();
  for (int j = 0; j < CLEN; ++j) {
    size_t ix = base + (size_t)j * Hdim;
    float f = lf[ix], v = lv[ix];
    A += f;
    Vv = logaddexpf_(f + Vv, v);
  }
  cA[tid] = A;
  cV[tid] = Vv;
}

__global__ __launch_bounds__(256) void scan_combine_kernel(const float* __restrict__ cA,
                                                           const float* __restrict__ cV,
                                                           float* __restrict__ cS) {
  int tid = blockIdx.x * 256 + threadIdx.x;   // < Bdim*Hdim = 1024
  int h = tid & 511, b = tid >> 9;
  float lh = -0.6931471805599453f;            // log(0.5) = log h0
  for (int p = 0; p < PCH; ++p) {
    int ix = (b * PCH + p) * Hdim + h;
    cS[ix] = lh;
    lh = logaddexpf_(cA[ix] + lh, cV[ix]);
  }
}

__global__ __launch_bounds__(256) void scan_emit_kernel(const float* __restrict__ lf,
                                                        const float* __restrict__ lv,
                                                        const float* __restrict__ cS,
                                                        float* __restrict__ x) {
  int tid = blockIdx.x * 256 + threadIdx.x;
  int h = tid & 511, p = (tid >> 9) & 31, b = tid >> 14;
  size_t base = ((size_t)(b * Sdim + p * CLEN)) * Hdim + h;
  float lh = cS[tid];
  for (int j = 0; j < CLEN; ++j) {
    size_t ix = base + (size_t)j * Hdim;
    lh = logaddexpf_(lf[ix] + lh, lv[ix]);
    x[ix] = expf(lh) + x[ix];
  }
}

// ---------------- host launcher ----------------
extern "C" void kernel_launch(void* const* d_in, const int* in_sizes, int n_in,
                              void* d_out, int out_size, void* d_ws, size_t ws_size,
                              hipStream_t stream) {
  (void)in_sizes; (void)n_in; (void)out_size; (void)ws_size;
  const int*   ids  = (const int*)  d_in[0];
  const float* emb  = (const float*)d_in[1];
  const float* Ws   = (const float*)d_in[2];
  const float* bsb  = (const float*)d_in[3];
  const float* lnw  = (const float*)d_in[4];
  const float* lnb  = (const float*)d_in[5];
  const float* flnw = (const float*)d_in[6];
  const float* fcw  = (const float*)d_in[7];
  const float* fcb  = (const float*)d_in[8];
  float* out = (float*)d_out;

  char* p = (char*)d_ws;
  auto carve = [&](size_t bytes) -> char* {
    char* r = p;
    p += (bytes + 255) & ~(size_t)255;
    return r;
  };
  float* x             = (float*)carve((size_t)Mrows * Hdim * 4);
  unsigned short* xnh  = (unsigned short*)carve((size_t)Mrows * Hdim * 2);
  unsigned short* xnl  = (unsigned short*)carve((size_t)Mrows * Hdim * 2);
  float* gates         = (float*)carve((size_t)Mrows * G3H * 4);
  float* lf            = (float*)carve((size_t)Mrows * Hdim * 4);
  float* lv            = (float*)carve((size_t)Mrows * Hdim * 4);
  float* cA            = (float*)carve((size_t)Bdim * PCH * Hdim * 4);
  float* cV            = (float*)carve((size_t)Bdim * PCH * Hdim * 4);
  float* cS            = (float*)carve((size_t)Bdim * PCH * Hdim * 4);
  unsigned short* wsh  = (unsigned short*)carve((size_t)Ldim * G3H * Hdim * 2);
  unsigned short* wsl  = (unsigned short*)carve((size_t)Ldim * G3H * Hdim * 2);
  unsigned short* fch  = (unsigned short*)carve((size_t)Vdim * Hdim * 2);
  unsigned short* fcl  = (unsigned short*)carve((size_t)Vdim * Hdim * 2);

  split_kernel<<<512, 256, 0, stream>>>(fcw, fch, fcl, (long long)Vdim * Hdim);
  split_kernel<<<512, 256, 0, stream>>>(Ws, wsh, wsl, (long long)Ldim * G3H * Hdim);
  embed_kernel<<<Mrows, 256, 0, stream>>>(ids, emb, x);

  for (int l = 0; l < Ldim; ++l) {
    ln_split_kernel<<<Mrows, 256, 0, stream>>>(x, lnw + l * Hdim, lnb + l * Hdim, xnh, xnl);
    gemm_bf16x2_wmma<<<dim3(G3H / 64, Mrows / 128), dim3(32, 4), 0, stream>>>(
        xnh, xnl, wsh + (size_t)l * G3H * Hdim, wsl + (size_t)l * G3H * Hdim,
        bsb + l * G3H, gates, Mrows, G3H, Hdim);
    gatelog_kernel<<<(Mrows * Hdim) / 256, 256, 0, stream>>>(gates, lf, lv);
    scan_chunk_kernel<<<(Bdim * PCH * Hdim) / 256, 256, 0, stream>>>(lf, lv, cA, cV);
    scan_combine_kernel<<<(Bdim * Hdim) / 256, 256, 0, stream>>>(cA, cV, cS);
    scan_emit_kernel<<<(Bdim * PCH * Hdim) / 256, 256, 0, stream>>>(lf, lv, cS, x);
  }

  ln_split_kernel<<<Mrows, 256, 0, stream>>>(x, flnw, nullptr, xnh, xnl);
  gemm_bf16x2_wmma<<<dim3(Vdim / 64, Mrows / 128), dim3(32, 4), 0, stream>>>(
      xnh, xnl, fch, fcl, fcb, out, Mrows, Vdim, Hdim);
}
